// MultiHeadAttention_57260503990948
// MI455X (gfx1250) — matrix operand order repaired and checked by reading, not compiled, not run on hardware
//
#include <hip/hip_runtime.h>
#include <hip/hip_bf16.h>
#include <math.h>

// Problem constants (match reference)
#define DM   1024          // model dim
#define NH   16            // heads
#define DK   64            // head dim
#define BB   4             // batch
#define SS   2048          // sequence
#define NTOK (BB * SS)     // 8192 tokens

typedef __attribute__((ext_vector_type(16))) __bf16 v16bf;
typedef __attribute__((ext_vector_type(4)))  __bf16 v4bf;
typedef __attribute__((ext_vector_type(8)))  float  v8f;
typedef __attribute__((ext_vector_type(4)))  unsigned int u32x4;
typedef __attribute__((ext_vector_type(8)))  int    i32x8;
typedef __attribute__((ext_vector_type(4)))  int    i32x4;

__device__ __forceinline__ v8f wmma_bf16(v16bf a, v16bf b, v8f c) {
    return __builtin_amdgcn_wmma_f32_16x16x32_bf16(
        false, a, false, b, (short)0, c, false, false);
}

// ---- Fragment builders (contiguous k order, same for A and B, wave32) ----
// A 16x32 bf16: lane L: row = L&15; half i -> K = kk + 16*(L>>4) + i (contiguous).
__device__ __forceinline__ v16bf a_frag_bf16(const __bf16* rowptr, int kk, int hl) {
    v16bf a;
    const __bf16* p0 = rowptr + kk + 16 * hl;
    const __bf16* p1 = rowptr + kk + 16 * hl + 8;
#pragma unroll
    for (int i = 0; i < 8; ++i) { a[i] = p0[i]; a[8 + i] = p1[i]; }
    return a;
}

// B 32x16 bf16 of W^T (B[k][n] = W[n][k], W row-major [N,K]):
// lane L: col n = n0 + (L&15); half i -> K = kk + 16*(L>>4) + i (contiguous).
__device__ __forceinline__ v16bf b_frag_bf16(const __bf16* W, int ld, int n0, int kk,
                                             int lr, int hl) {
    v16bf b;
    const __bf16* p = W + (size_t)(n0 + lr) * ld + kk + 16 * hl;
#pragma unroll
    for (int i = 0; i < 16; ++i) b[i] = p[i];
    return b;
}

__device__ __forceinline__ v16bf b_frag_lds(const __bf16* p) {
    v16bf b;
#pragma unroll
    for (int i = 0; i < 16; ++i) b[i] = p[i];
    return b;
}

// ---- Tensor Data Mover: 2D tile DMA (bf16) global -> LDS (ISA ch.8 D#) ----
__device__ __forceinline__ void tdm_load_2d_bf16(unsigned lds_off, const void* gaddr,
                                                 unsigned tensor_d0, unsigned tensor_d1,
                                                 unsigned stride0,
                                                 unsigned tile_d0, unsigned tile_d1) {
    const unsigned long long ga = (unsigned long long)(size_t)gaddr;
    u32x4 g0;
    g0[0] = 1u;                                   // count=1, no gather
    g0[1] = lds_off;                              // lds_addr (bytes)
    g0[2] = (unsigned)ga;                         // global_addr[31:0]
    g0[3] = (unsigned)((ga >> 32) & 0x01FFFFFFu)  // global_addr[56:32]
          | (2u << 30);                           // type=2 ("image")
    i32x8 g1;
    g1[0] = (int)(1u << 16);                      // wg_mask=0, data_size=1 (2B)
    g1[1] = (int)(tensor_d0 << 16);               // tensor_dim0[15:0] @ [31:16]
    g1[2] = (int)((tensor_d0 >> 16) | (tensor_d1 << 16));
    g1[3] = (int)((tensor_d1 >> 16) | (tile_d0 << 16));
    g1[4] = (int)(tile_d1 & 0xFFFFu);             // tile_dim1; tile_dim2=0
    g1[5] = (int)stride0;                         // tensor_dim0_stride[31:0]
    g1[6] = 0;
    g1[7] = 0;
    i32x4 z4 = {0, 0, 0, 0};
#if defined(__clang_major__) && (__clang_major__ >= 23)
    i32x8 z8 = {0, 0, 0, 0, 0, 0, 0, 0};
    __builtin_amdgcn_tensor_load_to_lds(g0, g1, z4, z4, z8, 0);
#else
    __builtin_amdgcn_tensor_load_to_lds(g0, g1, z4, z4, 0);
#endif
}

// ======================================================================
// Kernel 0: f32 -> bf16 conversion (vectorized), n4 = n/4
// ======================================================================
__global__ __launch_bounds__(256) void cvt_f32_to_bf16(const float* __restrict__ src,
                                                       __bf16* __restrict__ dst,
                                                       int n4) {
    const int i = blockIdx.x * blockDim.x + threadIdx.x;
    if (i >= n4) return;
    const float4 f = ((const float4*)src)[i];
    v4bf o;
    o[0] = (__bf16)f.x; o[1] = (__bf16)f.y; o[2] = (__bf16)f.z; o[3] = (__bf16)f.w;
    ((v4bf*)dst)[i] = o;
}

// ======================================================================
// Kernel 1: QKV projection  Y = X @ W^T + bias  -> bf16 [B,H,S,DK]
// All inputs bf16. One wave = 32x64 tile; 8 waves of a block share the N-tile.
// ======================================================================
__global__ __launch_bounds__(256) void proj_qkv(const __bf16* __restrict__ X,
                                                const __bf16* __restrict__ W,
                                                const float* __restrict__ bias,
                                                __bf16* __restrict__ Out) {
    const int lane = threadIdx.x & 31;
    const int lr = lane & 15, hl = lane >> 4;
    const int w = threadIdx.x >> 5;
    const int nt = blockIdx.x & 15;               // 16 col-tiles of 64
    const int mt = (blockIdx.x >> 4) * 8 + w;     // 256 row-tiles of 32
    if (mt >= NTOK / 32) return;
    const int m0 = mt * 32, n0 = nt * 64;

    const __bf16* xr0 = X + (size_t)(m0 + lr) * DM;
    const __bf16* xr1 = X + (size_t)(m0 + 16 + lr) * DM;

    v8f acc[2][4] = {};
    for (int kk = 0; kk < DM; kk += 32) {
        v16bf a0 = a_frag_bf16(xr0, kk, hl);
        v16bf a1 = a_frag_bf16(xr1, kk, hl);
#pragma unroll
        for (int t = 0; t < 4; ++t) {
            v16bf b = b_frag_bf16(W, DM, n0 + 16 * t, kk, lr, hl);
            acc[0][t] = wmma_bf16(a0, b, acc[0][t]);
            acc[1][t] = wmma_bf16(a1, b, acc[1][t]);
        }
    }
    // C layout: VGPR r -> row r + 8*hl, col = lane&15. Scatter to [B,H,S,DK].
#pragma unroll
    for (int i = 0; i < 2; ++i)
#pragma unroll
        for (int t = 0; t < 4; ++t) {
            const int col = n0 + 16 * t + lr;
            const int h = col >> 6, d = col & (DK - 1);
            const float bv = bias[col];
#pragma unroll
            for (int r = 0; r < 8; ++r) {
                const int row = m0 + 16 * i + r + 8 * hl;
                const int bb = row >> 11, s = row & (SS - 1);
                Out[(((size_t)(bb * NH + h)) * SS + s) * DK + d] =
                    (__bf16)(acc[i][t][r] + bv);
            }
        }
}

// ======================================================================
// Kernel 2: flash attention per (b,h). 8 waves x 16 q-rows = 128 q rows/block.
// K & V tiles streamed into double-buffered LDS by the Tensor Data Mover:
// wave 0 issues tile i+1's DMA before tile i is consumed, then retires
// tile i with s_wait_tensorcnt 2 (TENSORcnt completes in order per wave).
// ======================================================================
__global__ __launch_bounds__(256) void attn_kernel(const __bf16* __restrict__ Q,
                                                   const __bf16* __restrict__ K,
                                                   const __bf16* __restrict__ V,
                                                   __bf16* __restrict__ Aout) {
    __shared__ __bf16 Kt[2][64 * 64];     // K tile [kcol][d]  (2x8 KB, TDM dest)
    __shared__ __bf16 Vr[2][64 * 64];     // V tile [kcol][d]  (2x8 KB, TDM dest)
    __shared__ __bf16 Vt[64 * 64];        // V transposed [d][kcol] (8 KB)
    __shared__ __bf16 Pst[8][16 * 64];    // per-wave probs staging (16 KB)

    const int lane = threadIdx.x & 31;
    const int lr = lane & 15, hl = lane >> 4;
    const int w = threadIdx.x >> 5;
    const int bh = blockIdx.x >> 4;        // b*NH + h
    const int qt = blockIdx.x & 15;        // 16 q-tiles of 128 rows
    const int b = bh >> 4, h = bh & 15;
    const int q0 = qt * 128 + w * 16;

    const __bf16* Qh = Q + (size_t)bh * SS * DK;
    const __bf16* Kh = K + (size_t)bh * SS * DK;
    const __bf16* Vh = V + (size_t)bh * SS * DK;

    const unsigned ktOff0 = (unsigned)(size_t)(void*)&Kt[0][0];
    const unsigned ktOff1 = (unsigned)(size_t)(void*)&Kt[1][0];
    const unsigned vrOff0 = (unsigned)(size_t)(void*)&Vr[0][0];
    const unsigned vrOff1 = (unsigned)(size_t)(void*)&Vr[1][0];

    // Q fragments for this wave's 16 rows (K = 64 -> two 16x32 frags)
    const __bf16* qrow = Qh + (size_t)(q0 + lr) * DK;
    v16bf qf0 = a_frag_bf16(qrow, 0, hl);
    v16bf qf1 = a_frag_bf16(qrow, 32, hl);

    float rowM[8], rowL[8];
#pragma unroll
    for (int r = 0; r < 8; ++r) { rowM[r] = -INFINITY; rowL[r] = 0.f; }
    v8f o[4] = {};

    const int skr = threadIdx.x >> 2;          // 0..63 : k row
    const int sdq = (threadIdx.x & 3) * 16;    // 0,16,32,48 : d chunk

    const int NT = SS / 64;                    // 32 k-tiles

    // prologue: DMA tile 0 into buffer 0
    if (w == 0) {
        tdm_load_2d_bf16(ktOff0, Kh, DK, SS, DK, DK, 64);
        tdm_load_2d_bf16(vrOff0, Vh, DK, SS, DK, DK, 64);
    }

    for (int it = 0; it < NT; ++it) {
        const int cur = it & 1;
        __syncthreads();  // iter it-1 done reading buf[cur^1] (DMA target below)
        if (w == 0) {
            if (it + 1 < NT) {
                const size_t nb = (size_t)(it + 1) * 64 * DK;
                tdm_load_2d_bf16(cur ? ktOff0 : ktOff1, Kh + nb, DK, SS, DK, DK, 64);
                tdm_load_2d_bf16(cur ? vrOff0 : vrOff1, Vh + nb, DK, SS, DK, DK, 64);
                __builtin_amdgcn_s_wait_tensorcnt(2);  // retire tile it only
            } else {
                __builtin_amdgcn_s_wait_tensorcnt(0);
            }
        }
        __syncthreads();  // buf[cur] ready for whole block

        // transpose V tile in LDS: Vt[d][k] = Vr[cur][k][d]
        const __bf16* Vc = &Vr[cur][0];
#pragma unroll
        for (int i = 0; i < 16; ++i) Vt[(sdq + i) * 64 + skr] = Vc[skr * 64 + sdq + i];

        // scores: S = Q(16x64) @ K^T(64x64) -> four 16x16 f32 tiles
        const __bf16* Kc = &Kt[cur][0];
        v8f sc[4];
#pragma unroll
        for (int t = 0; t < 4; ++t) {
            v8f z = {};
            const __bf16* kp = &Kc[(16 * t + lr) * 64 + 16 * hl];
            z = wmma_bf16(qf0, b_frag_lds(kp), z);
            z = wmma_bf16(qf1, b_frag_lds(kp + 32), z);
            sc[t] = z;
        }
#pragma unroll
        for (int t = 0; t < 4; ++t)
#pragma unroll
            for (int r = 0; r < 8; ++r) sc[t][r] *= 0.125f;  // 1/sqrt(64)

        // online softmax (rows r+8*hl replicated across the 16-lane half)
#pragma unroll
        for (int r = 0; r < 8; ++r) {
            float m = fmaxf(fmaxf(sc[0][r], sc[1][r]), fmaxf(sc[2][r], sc[3][r]));
            m = fmaxf(m, __shfl_xor(m, 1, 32));
            m = fmaxf(m, __shfl_xor(m, 2, 32));
            m = fmaxf(m, __shfl_xor(m, 4, 32));
            m = fmaxf(m, __shfl_xor(m, 8, 32));
            const float newM = fmaxf(rowM[r], m);
            const float resc = __expf(rowM[r] - newM);
            rowM[r] = newM;
            float lsum = 0.f;
#pragma unroll
            for (int t = 0; t < 4; ++t) {
                const float p = __expf(sc[t][r] - newM);
                sc[t][r] = p;
                lsum += p;
            }
            lsum += __shfl_xor(lsum, 1, 32);
            lsum += __shfl_xor(lsum, 2, 32);
            lsum += __shfl_xor(lsum, 4, 32);
            lsum += __shfl_xor(lsum, 8, 32);
            rowL[r] = rowL[r] * resc + lsum;
            o[0][r] *= resc; o[1][r] *= resc; o[2][r] *= resc; o[3][r] *= resc;
        }

        // C-layout -> A-layout via per-wave LDS roundtrip (bf16 probs)
        __bf16* Pw = Pst[w];
#pragma unroll
        for (int t = 0; t < 4; ++t)
#pragma unroll
            for (int r = 0; r < 8; ++r)
                Pw[(r + 8 * hl) * 64 + 16 * t + lr] = (__bf16)sc[t][r];
        __syncthreads();  // Vt ready + Pst ready

        v16bf pf0 = a_frag_bf16(Pw + lr * 64, 0, hl);
        v16bf pf1 = a_frag_bf16(Pw + lr * 64, 32, hl);
#pragma unroll
        for (int td = 0; td < 4; ++td) {
            const __bf16* vp = &Vt[(16 * td + lr) * 64 + 16 * hl];
            o[td] = wmma_bf16(pf0, b_frag_lds(vp), o[td]);
            o[td] = wmma_bf16(pf1, b_frag_lds(vp + 32), o[td]);
        }
    }

    // normalize + store token-major [NTOK, DM] bf16 (col = h*64 + d)
#pragma unroll
    for (int td = 0; td < 4; ++td)
#pragma unroll
        for (int r = 0; r < 8; ++r) {
            const int qrowi = q0 + r + 8 * hl;
            const int token = b * SS + qrowi;
            const int col = h * DK + 16 * td + lr;
            Aout[(size_t)token * DM + col] = (__bf16)(o[td][r] / rowL[r]);
        }
}

// ======================================================================
// Kernel 3: output projection  out = attn @ W_o^T + b_o  (f32 output)
// ======================================================================
__global__ __launch_bounds__(256) void proj_out(const __bf16* __restrict__ X,
                                                const __bf16* __restrict__ W,
                                                const float* __restrict__ bias,
                                                float* __restrict__ Out) {
    const int lane = threadIdx.x & 31;
    const int lr = lane & 15, hl = lane >> 4;
    const int w = threadIdx.x >> 5;
    const int nt = blockIdx.x & 15;
    const int mt = (blockIdx.x >> 4) * 8 + w;
    if (mt >= NTOK / 32) return;
    const int m0 = mt * 32, n0 = nt * 64;

    const __bf16* xr0 = X + (size_t)(m0 + lr) * DM;
    const __bf16* xr1 = X + (size_t)(m0 + 16 + lr) * DM;

    v8f acc[2][4] = {};
    for (int kk = 0; kk < DM; kk += 32) {
        v16bf a0 = a_frag_bf16(xr0, kk, hl);
        v16bf a1 = a_frag_bf16(xr1, kk, hl);
#pragma unroll
        for (int t = 0; t < 4; ++t) {
            v16bf b = b_frag_bf16(W, DM, n0 + 16 * t, kk, lr, hl);
            acc[0][t] = wmma_bf16(a0, b, acc[0][t]);
            acc[1][t] = wmma_bf16(a1, b, acc[1][t]);
        }
    }
#pragma unroll
    for (int i = 0; i < 2; ++i)
#pragma unroll
        for (int t = 0; t < 4; ++t) {
            const int col = n0 + 16 * t + lr;
            const float bv = bias[col];
#pragma unroll
            for (int r = 0; r < 8; ++r) {
                const int row = m0 + 16 * i + r + 8 * hl;
                Out[(size_t)row * DM + col] = acc[i][t][r] + bv;
            }
        }
}

// ======================================================================
// Host launcher
// ======================================================================
extern "C" void kernel_launch(void* const* d_in, const int* in_sizes, int n_in,
                              void* d_out, int out_size, void* d_ws, size_t ws_size,
                              hipStream_t stream) {
    (void)in_sizes; (void)n_in; (void)out_size; (void)ws_size;
    const float* query = (const float*)d_in[0];
    const float* key_  = (const float*)d_in[1];
    const float* value = (const float*)d_in[2];
    const float* W_q = (const float*)d_in[3];
    const float* b_q = (const float*)d_in[4];
    const float* W_k = (const float*)d_in[5];
    const float* b_k = (const float*)d_in[6];
    const float* W_v = (const float*)d_in[7];
    const float* b_v = (const float*)d_in[8];
    const float* W_o = (const float*)d_in[9];
    const float* b_o = (const float*)d_in[10];

    char* ws = (char*)d_ws;
    const size_t actBytes = (size_t)NTOK * DM * sizeof(__bf16);   // 16 MiB
    __bf16* Xbf = (__bf16*)(ws + 0 * actBytes);  // activation staging; later Aws
    __bf16* Qws = (__bf16*)(ws + 1 * actBytes);
    __bf16* Kws = (__bf16*)(ws + 2 * actBytes);
    __bf16* Vws = (__bf16*)(ws + 3 * actBytes);
    __bf16* Wbf = (__bf16*)(ws + 4 * actBytes);  // weight staging (reused)
    __bf16* Aws = Xbf;                           // attn output reuses Xbf

    const int actN4 = NTOK * DM / 4;             // 2M float4 per activation
    const int wN4   = DM * DM / 4;               // 256K float4 per weight
    const dim3 tB(256);
    const dim3 gAct((actN4 + 255) / 256), gW((wN4 + 255) / 256);
    const dim3 gGemm(512);                       // 16 n-tiles x 32 m-groups

    // Q projection
    cvt_f32_to_bf16<<<gAct, tB, 0, stream>>>(query, Xbf, actN4);
    cvt_f32_to_bf16<<<gW,   tB, 0, stream>>>(W_q, Wbf, wN4);
    proj_qkv<<<gGemm, tB, 0, stream>>>(Xbf, Wbf, b_q, Qws);
    // K projection
    cvt_f32_to_bf16<<<gAct, tB, 0, stream>>>(key_, Xbf, actN4);
    cvt_f32_to_bf16<<<gW,   tB, 0, stream>>>(W_k, Wbf, wN4);
    proj_qkv<<<gGemm, tB, 0, stream>>>(Xbf, Wbf, b_k, Kws);
    // V projection
    cvt_f32_to_bf16<<<gAct, tB, 0, stream>>>(value, Xbf, actN4);
    cvt_f32_to_bf16<<<gW,   tB, 0, stream>>>(W_v, Wbf, wN4);
    proj_qkv<<<gGemm, tB, 0, stream>>>(Xbf, Wbf, b_v, Vws);

    // attention: B*H=64 head-slices x 16 q-tiles of 128 rows
    attn_kernel<<<dim3(BB * NH * (SS / 128)), tB, 0, stream>>>(Qws, Kws, Vws, Aws);

    // output projection
    cvt_f32_to_bf16<<<gW, tB, 0, stream>>>(W_o, Wbf, wN4);
    proj_out<<<gGemm, tB, 0, stream>>>(Aws, Wbf, b_o, (float*)d_out);
}
